// ATE_15496242004215
// MI455X (gfx1250) — matrix operand order repaired and checked by reading, not compiled, not run on hardware
//
#include <hip/hip_runtime.h>
#include <math.h>

// ---------------------------------------------------------------------------
// Problem constants
// ---------------------------------------------------------------------------
#define B_   32
#define L_   200
#define LP_  208         // L padded to 13 full 16-wide N tiles
#define KP_  224         // L padded to 7 full 32-deep K steps
#define DIM_ 36
#define D2_  72          // 2*DIM
#define ET_  128
#define NH_  64
#define SD_  9
#define NC_  2
#define NW_  288         // stacked W columns: [m1*x | m1 | m2*x | m2]

// output flat offsets (floats)
#define OFF_OUT_TE   0
#define OFF_OUT1_TE  147456      // 32*128*36
#define OFF_LOGITS   294912
#define OFF_SXG      294976
#define OFF_SOUT     336448
#define OFF_QP       377920

// workspace offsets (floats) for the f32 region
#define WS_QUERY_E   0            // 128*128
#define WS_KEY_E     16384        // 6400*128
#define WS_SCORES    835584       // 32*128*200
#define WS_O         1654784      // 32*128*288
#define WS_MEANX     2834432      // 32*72
#define WS_HFIN      2836736      // 32*64
#define WS_F32_END   2838784      // start of f16 region (32B aligned)

// offsets (halfs) inside the f16 region
#define HS_QH        0            // 128*128
#define HS_KH        16384        // 32*208*128
#define HS_EH        868352       // 32*128*224
#define HS_WH        1785856      // 32*288*224

typedef __attribute__((ext_vector_type(16))) _Float16 v16h;
typedef __attribute__((ext_vector_type(8)))  _Float16 v8h;
typedef __attribute__((ext_vector_type(8)))  float    v8f;

// ---------------------------------------------------------------------------
// 1a) time embeddings: key_e (6400x128), query_e (128x128)
// ---------------------------------------------------------------------------
__global__ void emb_kernel(const float* __restrict__ ts, const float* __restrict__ qp,
                           const float* __restrict__ pw, const float* __restrict__ pb,
                           const float* __restrict__ tw, const float* __restrict__ tb,
                           float* __restrict__ key_e, float* __restrict__ query_e) {
  int idx = blockIdx.x * blockDim.x + threadIdx.x;
  const int nkey = B_ * L_ * ET_;
  const int total = nkey + ET_ * ET_;
  if (idx >= total) return;
  float t; float* dst; int e;
  if (idx < nkey) {
    int row = idx / ET_; e = idx % ET_;
    t = ts[row]; dst = key_e + idx;
  } else {
    int i2 = idx - nkey;
    int row = i2 / ET_; e = i2 % ET_;
    t = qp[row]; dst = query_e + i2;
  }
  *dst = (e == 0) ? (t * tw[0] + tb[0]) : sinf(t * pw[e - 1] + pb[e - 1]);
}

// ---------------------------------------------------------------------------
// 1b) q = query_e @ qw^T + qb -> f16 ; k = key_e @ kw^T + kb -> f16, rows
//     zero-padded to 208 per batch so the WMMA N loop needs no guards.
// ---------------------------------------------------------------------------
__global__ void qk_kernel(const float* __restrict__ query_e, const float* __restrict__ key_e,
                          const float* __restrict__ qw, const float* __restrict__ qb,
                          const float* __restrict__ kw, const float* __restrict__ kb,
                          _Float16* __restrict__ qh, _Float16* __restrict__ kh) {
  int idx = blockIdx.x * blockDim.x + threadIdx.x;
  const int nq = ET_ * ET_;
  const int total = nq + B_ * LP_ * ET_;
  if (idx >= total) return;
  if (idx < nq) {
    int i = idx / ET_, j = idx % ET_;
    float s = qb[j];
    for (int e = 0; e < ET_; ++e) s += query_e[i * ET_ + e] * qw[j * ET_ + e];
    qh[idx] = (_Float16)s;
  } else {
    int i2 = idx - nq;
    int row = i2 / ET_, j = i2 % ET_;
    int b = row / LP_, l = row % LP_;
    float s = 0.f;
    if (l < L_) {
      s = kb[j];
      for (int e = 0; e < ET_; ++e) s += key_e[(b * L_ + l) * ET_ + e] * kw[j * ET_ + e];
    }
    kh[i2] = (_Float16)s;
  }
}

// ---------------------------------------------------------------------------
// WMMA fragment helpers (16-bit A 16x32, contiguous k order chosen here, not the ISA table's): per lane the A fragment
// is two contiguous 8-half runs: K = [16h .. 16h+7] and [16h+8 .. 16h+15].
// B fragment (K=e+16*half, n=lane&15) is one contiguous 16-half run.
// ---------------------------------------------------------------------------
__device__ __forceinline__ v16h load_a_frag(const _Float16* __restrict__ row,
                                            int kbase, int half) {
  v8h a0 = *(const v8h*)(row + kbase + 16 * half);
  v8h a1 = *(const v8h*)(row + kbase + 16 * half + 8);
  v16h a;
  #pragma unroll
  for (int e = 0; e < 8; ++e) { a[e] = a0[e]; a[e + 8] = a1[e]; }
  return a;
}

// ---------------------------------------------------------------------------
// 2) WMMA: scores[b] = q (128x128) @ k[b]^T (128x208pad) / sqrt(128)
// ---------------------------------------------------------------------------
__global__ void scores_kernel(const _Float16* __restrict__ qh, const _Float16* __restrict__ kh,
                              float* __restrict__ scores) {
  const int lane = threadIdx.x;
  const int tile = blockIdx.x;             // b*(8*13) + mt*13 + nt
  const int b  = tile / 104;
  const int r  = tile % 104;
  const int mt = r / 13;
  const int nt = r % 13;
  const int half = lane >> 4;
  const int mn = lane & 15;

  const _Float16* arow = qh + (mt * 16 + mn) * ET_;
  const _Float16* brow = kh + (b * LP_ + nt * 16 + mn) * ET_;

  v8f acc = {};
  #pragma unroll
  for (int kt = 0; kt < 4; ++kt) {
    v16h a  = load_a_frag(arow, kt * 32, half);
    v16h bf = *(const v16h*)(brow + kt * 32 + 16 * half);
    acc = __builtin_amdgcn_wmma_f32_16x16x32_f16(false, a, false, bf,
                                                 (short)0, acc, false, false);
  }
  const float scale = 0.08838834764831845f;        // 1/sqrt(128)
  const int lcol = nt * 16 + mn;
  if (lcol < L_) {
    #pragma unroll
    for (int r8 = 0; r8 < 8; ++r8) {
      int qrow = mt * 16 + r8 + 8 * half;
      scores[(b * ET_ + qrow) * L_ + lcol] = acc[r8] * scale;
    }
  }
}

// ---------------------------------------------------------------------------
// 3) E = exp(scores - rowmax), written f16 with stride 224 (zero padded)
// ---------------------------------------------------------------------------
__global__ void rowexp_kernel(const float* __restrict__ scores, _Float16* __restrict__ Eh) {
  int idx = blockIdx.x * blockDim.x + threadIdx.x;
  if (idx >= B_ * ET_) return;
  const float* row = scores + idx * L_;
  float m = -1e30f;
  for (int l = 0; l < L_; ++l) m = fmaxf(m, row[l]);
  _Float16* e = Eh + idx * KP_;
  for (int l = 0; l < L_; ++l) e[l] = (_Float16)expf(row[l] - m);
  for (int l = L_; l < KP_; ++l) e[l] = (_Float16)0.f;
}

// mean over l of x[b,l,d]  (fallback for fully-masked softmax)
__global__ void meanx_kernel(const float* __restrict__ x, float* __restrict__ meanx) {
  int idx = blockIdx.x * blockDim.x + threadIdx.x;
  if (idx >= B_ * D2_) return;
  int b = idx / D2_, d = idx % D2_;
  float s = 0.f;
  for (int l = 0; l < L_; ++l) s += x[(b * L_ + l) * D2_ + d];
  meanx[idx] = s * (1.0f / L_);
}

// ---------------------------------------------------------------------------
// 4a) pack W column-major f16: Wh[b][c][l], c in [0,288), l in [0,224) padded.
//     columns: [mask1*x | mask1 | mask2*x | mask2], mask1=obs, mask2=obs*rand
// ---------------------------------------------------------------------------
__global__ void packw_kernel(const float* __restrict__ x, const int* __restrict__ rmask,
                             _Float16* __restrict__ Wh) {
  int idx = blockIdx.x * blockDim.x + threadIdx.x;
  if (idx >= B_ * NW_ * KP_) return;
  int b = idx / (NW_ * KP_);
  int rem = idx % (NW_ * KP_);
  int c = rem / KP_, l = rem % KP_;
  float v = 0.f;
  if (l < L_) {
    int grp = c / D2_, d = c % D2_;
    float obs = x[(b * L_ + l) * D2_ + DIM_ + (d % DIM_)];
    float msk = (grp >= 2) ? obs * (float)rmask[b * L_ + l] : obs;
    v = (grp & 1) ? msk : msk * x[(b * L_ + l) * D2_ + d];
  }
  Wh[idx] = (_Float16)v;
}

// ---------------------------------------------------------------------------
// 4b) WMMA: O[b] = E[b] (128x224pad) @ W[b] (224x288)
// ---------------------------------------------------------------------------
__global__ void attn_mm_kernel(const _Float16* __restrict__ Eh, const _Float16* __restrict__ Wh,
                               float* __restrict__ O) {
  const int lane = threadIdx.x;
  const int tile = blockIdx.x;             // b*(8*18) + mt*18 + nt
  const int b  = tile / 144;
  const int r  = tile % 144;
  const int mt = r / 18;
  const int nt = r % 18;
  const int half = lane >> 4;
  const int mn = lane & 15;
  const int c  = nt * 16 + mn;             // 0..287, always in range

  const _Float16* arow = Eh + ((size_t)(b * ET_ + mt * 16 + mn)) * KP_;
  const _Float16* brow = Wh + ((size_t)(b * NW_ + c)) * KP_;

  v8f acc = {};
  #pragma unroll
  for (int kt = 0; kt < 7; ++kt) {
    v16h a  = load_a_frag(arow, kt * 32, half);
    v16h bf = *(const v16h*)(brow + kt * 32 + 16 * half);
    acc = __builtin_amdgcn_wmma_f32_16x16x32_f16(false, a, false, bf,
                                                 (short)0, acc, false, false);
  }
  #pragma unroll
  for (int r8 = 0; r8 < 8; ++r8) {
    int qrow = mt * 16 + r8 + 8 * half;
    O[(b * ET_ + qrow) * NW_ + c] = acc[r8];
  }
}

// ---------------------------------------------------------------------------
// 5) o = num/den (fallback meanx), then out = o @ ow^T + ob  (both masks)
// ---------------------------------------------------------------------------
__global__ void combine_project(const float* __restrict__ O, const float* __restrict__ meanx,
                                const float* __restrict__ ow, const float* __restrict__ ob,
                                float* __restrict__ out) {
  __shared__ float o1[D2_], o2[D2_];
  int b = blockIdx.x / ET_, qq = blockIdx.x % ET_;
  int t = threadIdx.x;                     // 96 threads
  if (t < D2_) {
    const float* Orow = O + (b * ET_ + qq) * NW_;
    float n1 = Orow[t],       d1 = Orow[72 + t];
    float n2 = Orow[144 + t], d2 = Orow[216 + t];
    float mx = meanx[b * D2_ + t];
    o1[t] = (d1 > 0.f) ? n1 / d1 : mx;
    o2[t] = (d2 > 0.f) ? n2 / d2 : mx;
  }
  __syncthreads();
  if (t < D2_) {
    int which = t / DIM_;                  // 0 -> out_te, 1 -> out1_te
    int j = t % DIM_;
    const float* src = which ? o2 : o1;
    float s = ob[j];
    for (int d = 0; d < D2_; ++d) s += src[d] * ow[j * D2_ + d];
    out[(which ? OFF_OUT1_TE : OFF_OUT_TE) + (b * ET_ + qq) * DIM_ + j] = s;
  }
}

// ---------------------------------------------------------------------------
// 6) sx_g = round(sigmoid(sx^T sx)) ; 7) sout = sigmoid(out_te^T out_te)
// ---------------------------------------------------------------------------
__global__ void sxg_kernel(const float* __restrict__ x, float* __restrict__ out) {
  int idx = blockIdx.x * blockDim.x + threadIdx.x;
  if (idx >= B_ * DIM_ * DIM_) return;
  int b = idx / (DIM_ * DIM_), de = idx % (DIM_ * DIM_);
  int d = de / DIM_, e = de % DIM_;
  float s = 0.f;
  for (int l = 0; l < L_; ++l)
    s += x[(b * L_ + l) * D2_ + d] * x[(b * L_ + l) * D2_ + e];
  out[OFF_SXG + idx] = rintf(1.f / (1.f + expf(-s)));
}

__global__ void sout_kernel(const float* __restrict__ out_te, float* __restrict__ out) {
  int idx = blockIdx.x * blockDim.x + threadIdx.x;
  if (idx >= B_ * DIM_ * DIM_) return;
  int b = idx / (DIM_ * DIM_), de = idx % (DIM_ * DIM_);
  int d = de / DIM_, e = de % DIM_;
  float s = 0.f;
  for (int qq = 0; qq < ET_; ++qq)
    s += out_te[(b * ET_ + qq) * DIM_ + d] * out_te[(b * ET_ + qq) * DIM_ + e];
  out[OFF_SOUT + idx] = 1.f / (1.f + expf(-s));
}

// ---------------------------------------------------------------------------
// 8) GRU scan over the 128 query steps; one block per batch, 192 threads
// ---------------------------------------------------------------------------
__global__ void gru_kernel(const float* __restrict__ out_te,
                           const float* __restrict__ wih, const float* __restrict__ whh,
                           const float* __restrict__ bih, const float* __restrict__ bhh,
                           float* __restrict__ hfin) {
  __shared__ float h[NH_], xt[DIM_], gi[3 * NH_], gh[3 * NH_], hn[NH_];
  int b = blockIdx.x, t = threadIdx.x;     // t in [0,192)
  if (t < NH_) h[t] = 0.f;
  __syncthreads();
  for (int qq = 0; qq < ET_; ++qq) {
    if (t < DIM_) xt[t] = out_te[(b * ET_ + qq) * DIM_ + t];
    __syncthreads();
    float sgi = bih[t];
    for (int d = 0; d < DIM_; ++d) sgi += xt[d] * wih[t * DIM_ + d];
    float sgh = bhh[t];
    for (int j = 0; j < NH_; ++j) sgh += h[j] * whh[t * NH_ + j];
    gi[t] = sgi; gh[t] = sgh;
    __syncthreads();
    if (t < NH_) {
      float rr = 1.f / (1.f + expf(-(gi[t] + gh[t])));
      float zz = 1.f / (1.f + expf(-(gi[NH_ + t] + gh[NH_ + t])));
      float nn = tanhf(gi[2 * NH_ + t] + rr * gh[2 * NH_ + t]);
      hn[t] = (1.f - zz) * nn + zz * h[t];
    }
    __syncthreads();
    if (t < NH_) h[t] = hn[t];
    __syncthreads();
  }
  if (t < NH_) hfin[b * NH_ + t] = h[t];
}

// ---------------------------------------------------------------------------
// 9) classifier head + batchnorm + exact GELU + query_param copy (1 block)
// ---------------------------------------------------------------------------
__global__ void cls_kernel(const float* __restrict__ hfin, const float* __restrict__ sinfo,
                           const float* __restrict__ stw, const float* __restrict__ stb,
                           const float* __restrict__ c1w, const float* __restrict__ c1b,
                           const float* __restrict__ bng, const float* __restrict__ bnb,
                           const float* __restrict__ c2w, const float* __restrict__ c2b,
                           const float* __restrict__ qp, float* __restrict__ out) {
  __shared__ float cin[B_ * D2_], z1[B_ * D2_], zg[B_ * D2_];
  __shared__ float mu[D2_], var_[D2_];
  int t = threadIdx.x;
  for (int i = t; i < B_ * D2_; i += blockDim.x) {
    int b = i / D2_, j = i % D2_;
    float v;
    if (j < NH_) v = hfin[b * NH_ + j];
    else {
      int s8 = j - NH_;
      float s = stb[s8];
      for (int d = 0; d < SD_; ++d) s += sinfo[b * SD_ + d] * stw[s8 * SD_ + d];
      v = s;
    }
    cin[i] = v;
  }
  __syncthreads();
  for (int i = t; i < B_ * D2_; i += blockDim.x) {
    int b = i / D2_, j = i % D2_;
    float s = c1b[j];
    for (int d = 0; d < D2_; ++d) s += cin[b * D2_ + d] * c1w[j * D2_ + d];
    z1[i] = s;
  }
  __syncthreads();
  for (int j = t; j < D2_; j += blockDim.x) {
    float m = 0.f;
    for (int b = 0; b < B_; ++b) m += z1[b * D2_ + j];
    m *= (1.f / B_);
    float v = 0.f;
    for (int b = 0; b < B_; ++b) { float d = z1[b * D2_ + j] - m; v += d * d; }
    mu[j] = m; var_[j] = v * (1.f / B_);
  }
  __syncthreads();
  for (int i = t; i < B_ * D2_; i += blockDim.x) {
    int j = i % D2_;
    float zn = (z1[i] - mu[j]) * rsqrtf(var_[j] + 1e-5f) * bng[j] + bnb[j];
    zg[i] = 0.5f * zn * (1.f + erff(zn * 0.7071067811865476f));
  }
  __syncthreads();
  for (int i = t; i < B_ * NC_; i += blockDim.x) {
    int b = i / NC_, c = i % NC_;
    float s = c2b[c];
    for (int j = 0; j < D2_; ++j) s += zg[b * D2_ + j] * c2w[c * D2_ + j];
    out[OFF_LOGITS + i] = s;
  }
  for (int i = t; i < ET_; i += blockDim.x) out[OFF_QP + i] = qp[i];
}

// ---------------------------------------------------------------------------
extern "C" void kernel_launch(void* const* d_in, const int* in_sizes, int n_in,
                              void* d_out, int out_size, void* d_ws, size_t ws_size,
                              hipStream_t stream) {
  const float* x      = (const float*)d_in[0];
  const float* ts     = (const float*)d_in[1];
  const float* sinfo  = (const float*)d_in[2];
  const float* qp     = (const float*)d_in[3];
  const float* pw     = (const float*)d_in[4];
  const float* pb     = (const float*)d_in[5];
  const float* tw     = (const float*)d_in[6];
  const float* tb     = (const float*)d_in[7];
  const float* qw     = (const float*)d_in[8];
  const float* qb     = (const float*)d_in[9];
  const float* kw     = (const float*)d_in[10];
  const float* kb     = (const float*)d_in[11];
  const float* ow     = (const float*)d_in[12];
  const float* ob     = (const float*)d_in[13];
  const float* wih    = (const float*)d_in[14];
  const float* whh    = (const float*)d_in[15];
  const float* bih    = (const float*)d_in[16];
  const float* bhh    = (const float*)d_in[17];
  const float* stw    = (const float*)d_in[18];
  const float* stb    = (const float*)d_in[19];
  const float* c1w    = (const float*)d_in[20];
  const float* c1b    = (const float*)d_in[21];
  const float* bng    = (const float*)d_in[22];
  const float* bnb    = (const float*)d_in[23];
  const float* c2w    = (const float*)d_in[24];
  const float* c2b    = (const float*)d_in[25];
  const int*   rmask  = (const int*)d_in[26];

  float* out = (float*)d_out;
  float* ws  = (float*)d_ws;
  float* query_e = ws + WS_QUERY_E;
  float* key_e   = ws + WS_KEY_E;
  float* scores  = ws + WS_SCORES;
  float* Obuf    = ws + WS_O;
  float* meanx   = ws + WS_MEANX;
  float* hfin    = ws + WS_HFIN;
  _Float16* hbase = (_Float16*)(ws + WS_F32_END);
  _Float16* qh = hbase + HS_QH;
  _Float16* kh = hbase + HS_KH;
  _Float16* Eh = hbase + HS_EH;
  _Float16* Wh = hbase + HS_WH;

  const int nEmb = B_ * L_ * ET_ + ET_ * ET_;
  emb_kernel<<<(nEmb + 255) / 256, 256, 0, stream>>>(ts, qp, pw, pb, tw, tb, key_e, query_e);
  const int nQK = ET_ * ET_ + B_ * LP_ * ET_;
  qk_kernel<<<(nQK + 255) / 256, 256, 0, stream>>>(query_e, key_e, qw, qb, kw, kb, qh, kh);

  scores_kernel<<<B_ * 8 * 13, 32, 0, stream>>>(qh, kh, scores);
  rowexp_kernel<<<(B_ * ET_ + 255) / 256, 256, 0, stream>>>(scores, Eh);
  meanx_kernel<<<(B_ * D2_ + 255) / 256, 256, 0, stream>>>(x, meanx);
  packw_kernel<<<(B_ * NW_ * KP_ + 255) / 256, 256, 0, stream>>>(x, rmask, Wh);
  attn_mm_kernel<<<B_ * 8 * 18, 32, 0, stream>>>(Eh, Wh, Obuf);
  combine_project<<<B_ * ET_, 96, 0, stream>>>(Obuf, meanx, ow, ob, out);

  sxg_kernel<<<(B_ * DIM_ * DIM_ + 255) / 256, 256, 0, stream>>>(x, out);
  sout_kernel<<<(B_ * DIM_ * DIM_ + 255) / 256, 256, 0, stream>>>(out + OFF_OUT_TE, out);
  gru_kernel<<<B_, 3 * NH_, 0, stream>>>(out + OFF_OUT_TE, wih, whh, bih, bhh, hfin);
  cls_kernel<<<1, 128, 0, stream>>>(hfin, sinfo, stw, stb, c1w, c1b, bng, bnb, c2w, c2b, qp, out);
}
